// WindowAttention_12953621364985
// MI455X (gfx1250) — matrix operand order repaired and checked by reading, not compiled, not run on hardware
//
#include <hip/hip_runtime.h>

typedef __attribute__((ext_vector_type(8)))  float  v8f;
typedef __attribute__((ext_vector_type(16))) __bf16 v16bf;
typedef __attribute__((ext_vector_type(8)))  __bf16 v8bf;

#define NHEADS 12
#define CCH    768

// ---------------------------------------------------------------------------
// WMMA wrapper: D = A(16x32 bf16) * B(32x16 bf16) + C(16x16 f32)
// ---------------------------------------------------------------------------
__device__ __forceinline__ v8f wmma_bf16(v16bf a, v16bf b, v8f c) {
  return __builtin_amdgcn_wmma_f32_16x16x32_bf16(
      /*neg_a=*/false, a, /*neg_b=*/false, b,
      /*c_mod=*/(short)0, c, /*reuse_a=*/false, /*reuse_b=*/false);
}

// ---------------------------------------------------------------------------
// Fragment loaders (CDNA5 ISA 7.12.2, wave32)
// A (16x32, 16-bit): lane L -> row m0+(L&15); elems 0..7 = K[k0+8*hi .. +7],
//                    elems 8..15 = K[k0+8*hi+16 .. +23], hi = (L>=16)
// B (32x16, 16-bit): lane L -> col n0+(L&15); elems 0..7 = K[k0+8*hi .. +7],
//                    elems 8..15 = K[k0+8*hi+16 .. +23]
// ---------------------------------------------------------------------------
__device__ __forceinline__ v16bf fragA_bf(const __bf16* __restrict__ s, int ld,
                                          int m0, int k0, int lane) {
  const __bf16* p = s + (size_t)(m0 + (lane & 15)) * ld + k0 + ((lane >> 4) << 3);
  v8bf lo = *(const v8bf*)p;
  v8bf hi = *(const v8bf*)(p + 16);
  return __builtin_shufflevector(lo, hi, 0, 1, 2, 3, 4, 5, 6, 7,
                                 8, 9, 10, 11, 12, 13, 14, 15);
}

__device__ __forceinline__ v16bf fragB_bf(const __bf16* __restrict__ s, int ld,
                                          int n0, int k0, int lane) {
  const __bf16* p = s + (size_t)(n0 + (lane & 15)) * ld + k0 + ((lane >> 4) << 3);
  v8bf lo = *(const v8bf*)p;
  v8bf hi = *(const v8bf*)(p + 16);
  return __builtin_shufflevector(lo, hi, 0, 1, 2, 3, 4, 5, 6, 7,
                                 8, 9, 10, 11, 12, 13, 14, 15);
}

// A fragment: global bf16 X rows gathered through rowOff (element offsets)
__device__ __forceinline__ v16bf fragA_xbf(const __bf16* __restrict__ s,
                                           const int* rowOff, int m0, int k0,
                                           int lane) {
  const __bf16* p = s + rowOff[m0 + (lane & 15)] + k0 + ((lane >> 4) << 3);
  v8bf lo = *(const v8bf*)p;
  v8bf hi = *(const v8bf*)(p + 16);
  return __builtin_shufflevector(lo, hi, 0, 1, 2, 3, 4, 5, 6, 7,
                                 8, 9, 10, 11, 12, 13, 14, 15);
}

// ---- fallback loaders (f32 source, hardware cvt via static_cast) ----------
__device__ __forceinline__ v16bf fragA_xf32(const float* __restrict__ x,
                                            const int* rowOff, int m0, int k0,
                                            int lane) {
  const float* p = x + rowOff[m0 + (lane & 15)] + k0 + ((lane >> 4) << 3);
  v16bf f;
#pragma unroll
  for (int i = 0; i < 8; ++i) f[i] = (__bf16)p[i];
#pragma unroll
  for (int i = 0; i < 8; ++i) f[8 + i] = (__bf16)p[16 + i];
  return f;
}

__device__ __forceinline__ v16bf fragB_wf32(const float* __restrict__ w, int ld,
                                            int n0, int k0, int lane) {
  const float* p = w + (size_t)(n0 + (lane & 15)) * ld + k0 + ((lane >> 4) << 3);
  v16bf f;
#pragma unroll
  for (int i = 0; i < 8; ++i) f[i] = (__bf16)p[i];
#pragma unroll
  for (int i = 0; i < 8; ++i) f[8 + i] = (__bf16)p[16 + i];
  return f;
}

// D stores. C/D layout: lane L -> col n0+(L&15); VGPR i -> row m0+8*hi+i
__device__ __forceinline__ void storeD_lds(__bf16* s, int ld, int m0, int n0,
                                           v8f c, int lane) {
  int n = n0 + (lane & 15);
  int mb = m0 + ((lane >> 4) << 3);
#pragma unroll
  for (int i = 0; i < 8; ++i) s[(mb + i) * ld + n] = (__bf16)c[i];
}

// transposed store (for V^T): row = channel, 8 contiguous token columns
__device__ __forceinline__ void storeD_ldsT(__bf16* s, int ld, int m0, int n0,
                                            v8f c, int lane) {
  __bf16* p = s + (n0 + (lane & 15)) * ld + m0 + ((lane >> 4) << 3);
#pragma unroll
  for (int i = 0; i < 8; ++i) p[i] = (__bf16)c[i];
}

// ---------------------------------------------------------------------------
// f32 -> bf16 bulk conversion (memory-bound, one pass over x / weights)
// ---------------------------------------------------------------------------
__global__ __launch_bounds__(256) void cvt_f32_to_bf16_kernel(
    const float* __restrict__ in, __bf16* __restrict__ out, int n) {
  int stride = gridDim.x * blockDim.x * 4;
  for (int i = (blockIdx.x * blockDim.x + threadIdx.x) * 4; i < n; i += stride) {
    float4 v = *(const float4*)(in + i);
    out[i + 0] = (__bf16)v.x;
    out[i + 1] = (__bf16)v.y;
    out[i + 2] = (__bf16)v.z;
    out[i + 3] = (__bf16)v.w;
  }
}

// ---------------------------------------------------------------------------
// Fused window attention: one 256-thread block (8 waves) per 8x8 window.
// ---------------------------------------------------------------------------
template <bool PRE>
__global__ __launch_bounds__(256) void WindowAttention_fused_kernel(
    const float* __restrict__ x, const __bf16* __restrict__ xbf,
    const float* __restrict__ qkv_w, const __bf16* __restrict__ qkv_wbf,
    const float* __restrict__ qkv_b,
    const float* __restrict__ proj_w, const __bf16* __restrict__ proj_wbf,
    const float* __restrict__ proj_b, float* __restrict__ out) {
  __shared__ __align__(16) __bf16 sQ[64 * 64];
  __shared__ __align__(16) __bf16 sK[64 * 64];
  __shared__ __align__(16) __bf16 sVt[64 * 64];   // V transposed: [d][token]
  __shared__ __align__(16) __bf16 sP[64 * 64];    // softmax probs
  __shared__ __align__(16) __bf16 sO[64 * CCH];   // concatenated head outputs
  __shared__ __align__(16) float  sS[64 * 64];    // f32 logits
  __shared__ int rowOff[64];                      // token -> global row offset

  const int tid = threadIdx.x;
  const int lane = tid & 31;
  const int wave = tid >> 5;

  const int wb = blockIdx.x;          // window id [0, 512)
  const int b = wb >> 6;
  const int wr = (wb & 63) >> 3;
  const int wc = wb & 7;

  if (tid < 64) {
    int t = tid;
    int gy = wr * 8 + (t >> 3);
    int gx = wc * 8 + (t & 7);
    rowOff[t] = (b * 4096 + gy * 64 + gx) * CCH;
  }
  __syncthreads();

  // wave -> (m-tile, half): each wave keeps 6 accumulators so the X
  // A-fragment is loaded once per k-step and feeds 6 WMMAs.
  const int m0q = (wave >> 1) * 16;   // QKV phase M tile
  const int half = wave & 1;          // 6 col-tiles out of 12 (Q4,K4,V4)

  for (int head = 0; head < NHEADS; ++head) {
    // -------- QKV projection for this head --------
    {
      v8f acc[6];
#pragma unroll
      for (int j = 0; j < 6; ++j) acc[j] = (v8f){0.f,0.f,0.f,0.f,0.f,0.f,0.f,0.f};
#pragma unroll 2
      for (int k0 = 0; k0 < CCH; k0 += 32) {
        v16bf a;
        if constexpr (PRE) a = fragA_xbf(xbf, rowOff, m0q, k0, lane);
        else               a = fragA_xf32(x, rowOff, m0q, k0, lane);
#pragma unroll
        for (int j = 0; j < 6; ++j) {
          int c = half * 6 + j;                       // 0..11
          int wrow0 = (c >> 2) * CCH + head * 64 + (c & 3) * 16;
          v16bf bf;
          if constexpr (PRE) bf = fragB_bf(qkv_wbf, CCH, wrow0, k0, lane);
          else               bf = fragB_wf32(qkv_w, CCH, wrow0, k0, lane);
          acc[j] = wmma_bf16(a, bf, acc[j]);
        }
      }
#pragma unroll
      for (int j = 0; j < 6; ++j) {
        int c = half * 6 + j;
        int mat = c >> 2;
        int n0 = (c & 3) * 16;
        int wrow0 = mat * CCH + head * 64 + n0;
        float bias = qkv_b[wrow0 + (lane & 15)];
#pragma unroll
        for (int i = 0; i < 8; ++i) acc[j][i] += bias;
        if (mat == 0)      storeD_lds (sQ, 64, m0q, n0, acc[j], lane);
        else if (mat == 1) storeD_lds (sK, 64, m0q, n0, acc[j], lane);
        else               storeD_ldsT(sVt, 64, m0q, n0, acc[j], lane);
      }
    }
    __syncthreads();

    // -------- S = Q * K^T --------
    for (int tile = wave; tile < 16; tile += 8) {
      int m0 = (tile >> 2) * 16;
      int n0 = (tile & 3) * 16;
      v8f acc = {0.f, 0.f, 0.f, 0.f, 0.f, 0.f, 0.f, 0.f};
#pragma unroll
      for (int k0 = 0; k0 < 64; k0 += 32) {
        v16bf a = fragA_bf(sQ, 64, m0, k0, lane);
        v16bf bf = fragB_bf(sK, 64, n0, k0, lane);
        acc = wmma_bf16(a, bf, acc);
      }
      int n = n0 + (lane & 15);
      int mb = m0 + ((lane >> 4) << 3);
#pragma unroll
      for (int i = 0; i < 8; ++i) sS[(mb + i) * 64 + n] = acc[i];
    }
    __syncthreads();

    // -------- softmax (scale folded into exp): 4 lanes per row --------
    {
      const float scale = 0.125f;  // hd^-0.5 = 1/8
      int r = tid >> 2;
      int q4 = tid & 3;
      float vals[16];
      float mx = -3.4e38f;
#pragma unroll
      for (int i = 0; i < 16; ++i) {
        vals[i] = sS[r * 64 + q4 * 16 + i];
        mx = fmaxf(mx, vals[i]);
      }
      mx = fmaxf(mx, __shfl_xor(mx, 1, 32));
      mx = fmaxf(mx, __shfl_xor(mx, 2, 32));
      float sum = 0.f;
#pragma unroll
      for (int i = 0; i < 16; ++i) {
        vals[i] = __expf((vals[i] - mx) * scale);
        sum += vals[i];
      }
      sum += __shfl_xor(sum, 1, 32);
      sum += __shfl_xor(sum, 2, 32);
      float inv = 1.f / sum;
#pragma unroll
      for (int i = 0; i < 16; ++i)
        sP[r * 64 + q4 * 16 + i] = (__bf16)(vals[i] * inv);
    }
    __syncthreads();

    // -------- O_head = P * V --------
    for (int tile = wave; tile < 16; tile += 8) {
      int m0 = (tile >> 2) * 16;
      int n0 = (tile & 3) * 16;
      v8f acc = {0.f, 0.f, 0.f, 0.f, 0.f, 0.f, 0.f, 0.f};
#pragma unroll
      for (int k0 = 0; k0 < 64; k0 += 32) {
        v16bf a = fragA_bf(sP, 64, m0, k0, lane);
        v16bf bf = fragB_bf(sVt, 64, n0, k0, lane);
        acc = wmma_bf16(a, bf, acc);
      }
      storeD_lds(sO, CCH, m0, head * 64 + n0, acc, lane);
    }
    __syncthreads();  // protect per-head buffers before next iteration
  }

  // -------- final projection: out = O(64x768) * proj_w^T + proj_b --------
  for (int tile = wave; tile < 192; tile += 8) {
    int m0 = (tile / 48) * 16;
    int n0 = (tile % 48) * 16;
    v8f acc = {0.f, 0.f, 0.f, 0.f, 0.f, 0.f, 0.f, 0.f};
#pragma unroll 2
    for (int k0 = 0; k0 < CCH; k0 += 32) {
      v16bf a = fragA_bf(sO, CCH, m0, k0, lane);
      v16bf bf;
      if constexpr (PRE) bf = fragB_bf(proj_wbf, CCH, n0, k0, lane);
      else               bf = fragB_wf32(proj_w, CCH, n0, k0, lane);
      acc = wmma_bf16(a, bf, acc);
    }
    int n = n0 + (lane & 15);
    float bias = proj_b[n];
    int mb = m0 + ((lane >> 4) << 3);
#pragma unroll
    for (int i = 0; i < 8; ++i) out[rowOff[mb + i] + n] = acc[i] + bias;
  }
}

extern "C" void kernel_launch(void* const* d_in, const int* in_sizes, int n_in,
                              void* d_out, int out_size, void* d_ws,
                              size_t ws_size, hipStream_t stream) {
  const float* x      = (const float*)d_in[0];
  const float* qkv_w  = (const float*)d_in[1];
  const float* qkv_b  = (const float*)d_in[2];
  const float* proj_w = (const float*)d_in[3];
  const float* proj_b = (const float*)d_in[4];
  float* out = (float*)d_out;

  const size_t nX  = (size_t)8 * 4096 * 768;   // 25,165,824
  const size_t nQW = (size_t)2304 * 768;       //  1,769,472
  const size_t nPW = (size_t)768 * 768;        //    589,824
  const size_t need = (nX + nQW + nPW) * 2;    // ~52.5 MB of bf16

  if (ws_size >= need) {
    __bf16* xbf  = (__bf16*)d_ws;
    __bf16* qwbf = xbf + nX;
    __bf16* pwbf = qwbf + nQW;
    cvt_f32_to_bf16_kernel<<<4096, 256, 0, stream>>>(x, xbf, (int)nX);
    cvt_f32_to_bf16_kernel<<<1024, 256, 0, stream>>>(qkv_w, qwbf, (int)nQW);
    cvt_f32_to_bf16_kernel<<<512, 256, 0, stream>>>(proj_w, pwbf, (int)nPW);
    WindowAttention_fused_kernel<true><<<dim3(512), dim3(256), 0, stream>>>(
        x, xbf, qkv_w, qwbf, qkv_b, proj_w, pwbf, proj_b, out);
  } else {
    WindowAttention_fused_kernel<false><<<dim3(512), dim3(256), 0, stream>>>(
        x, nullptr, qkv_w, nullptr, qkv_b, proj_w, nullptr, proj_b, out);
  }
}
